// SelfAttention2d_46042049413891
// MI455X (gfx1250) — compile-verified, matrix operand order checked by reading, not run on hardware
//
#include <hip/hip_runtime.h>
#include <hip/hip_bf16.h>

// MI455X / gfx1250 self-attention-2d: WMMA f16 matmuls + async-to-LDS staging.

typedef _Float16 half_t;
typedef __attribute__((ext_vector_type(8)))  _Float16 v8h;
typedef __attribute__((ext_vector_type(16))) _Float16 v16h;
typedef __attribute__((ext_vector_type(8)))  float    v8f;
typedef __attribute__((ext_vector_type(4)))  int      v4i;

#define LCNT 1024   // 32*32 spatial positions
#define CH   256
#define NH   4
#define HD   64

#if __has_builtin(__builtin_amdgcn_global_load_async_to_lds_b128)
#define HAS_ASYNC 1
#else
#define HAS_ASYNC 0
#endif

#if HAS_ASYNC
typedef __attribute__((address_space(1))) v4i gv4i_t;   // global int4
typedef __attribute__((address_space(3))) v4i lv4i_t;   // LDS int4
#endif

// One 16-byte global->LDS copy per lane. Async (ASYNCcnt) when available.
__device__ __forceinline__ void async_b128(const half_t* g, half_t* l) {
#if HAS_ASYNC
  __builtin_amdgcn_global_load_async_to_lds_b128((gv4i_t*)g, (lv4i_t*)l, 0, 0);
#else
  *(v8h*)l = *(const v8h*)g;   // sync fallback: global load + ds store
#endif
}

template <int N>
__device__ __forceinline__ void wait_async_le() {
#if HAS_ASYNC
#if __has_builtin(__builtin_amdgcn_s_wait_asynccnt)
  __builtin_amdgcn_s_wait_asynccnt(N);
#else
  asm volatile("s_wait_asynccnt %0" :: "i"(N) : "memory");
#endif
#endif
}

// 16x32 f16 fragment (A-layout, ISA 7.12.2) from a row-major matrix with K as
// the fast dim. Lanes 0-15: row M=lane, K in {0..7,16..23}; lanes 16-31:
// K in {8..15,24..31}. Also used for B stored as B^T (row = N, col = K).
__device__ __forceinline__ v16h load_frag(const half_t* __restrict__ base,
                                          int ld, int row0, int k0) {
  int lane = threadIdx.x & 31;
  const half_t* p = base + (size_t)(row0 + (lane & 15)) * ld + k0 + ((lane >> 4) << 3);
  v8h lo = *(const v8h*)(p);
  v8h hi = *(const v8h*)(p + 16);
  v16h f;
#pragma unroll
  for (int i = 0; i < 8; ++i) { f[i] = lo[i]; f[8 + i] = hi[i]; }
  return f;
}

__device__ __forceinline__ v8f wmma_f16(v16h a, v16h b, v8f c) {
  return __builtin_amdgcn_wmma_f32_16x16x32_f16(false, a, false, b, (short)0, c,
                                                false, false);
}

// ---------------- weight fp32 -> fp16 ----------------
__global__ __launch_bounds__(256) void cvt_kernel(const float* __restrict__ qkv_w,
                                                  const float* __restrict__ proj_w,
                                                  half_t* __restrict__ wq,
                                                  half_t* __restrict__ wp) {
  int i = blockIdx.x * 256 + threadIdx.x;
  if (i < 3 * CH * CH) wq[i] = (half_t)qkv_w[i];
  if (i < CH * CH)     wp[i] = (half_t)proj_w[i];
}

// ---------------- GroupNorm -> transposed y_t[b][l][c] f16 ----------------
__global__ __launch_bounds__(256) void gn_kernel(const float* __restrict__ x,
                                                 const float* __restrict__ gw,
                                                 const float* __restrict__ gb,
                                                 half_t* __restrict__ y_t) {
  const int CPG = 8;
  int b = blockIdx.x >> 5;
  int g = blockIdx.x & 31;
  const float* xg = x + ((size_t)b * CH + g * CPG) * LCNT;
  float s = 0.f, s2 = 0.f;
  for (int idx = threadIdx.x; idx < CPG * LCNT; idx += 256) {
    float v = xg[idx];
    s += v; s2 += v * v;
  }
#pragma unroll
  for (int off = 16; off > 0; off >>= 1) {
    s  += __shfl_xor(s, off, 32);
    s2 += __shfl_xor(s2, off, 32);
  }
  __shared__ float rs[8], rq[8];
  int wave = threadIdx.x >> 5, lane = threadIdx.x & 31;
  if (lane == 0) { rs[wave] = s; rq[wave] = s2; }
  __syncthreads();
  s = 0.f; s2 = 0.f;
#pragma unroll
  for (int i = 0; i < 8; ++i) { s += rs[i]; s2 += rq[i]; }
  const float invN = 1.0f / (float)(CPG * LCNT);
  float mean = s * invN;
  float var  = s2 * invN - mean * mean;
  float rstd = rsqrtf(var + 1e-5f);
  for (int idx = threadIdx.x; idx < CPG * LCNT; idx += 256) {
    int cl = idx >> 10;
    int l  = idx & (LCNT - 1);
    int c  = g * CPG + cl;
    float v = (xg[idx] - mean) * rstd * gw[c] + gb[c];
    y_t[((size_t)b * LCNT + l) * CH + c] = (half_t)v;
  }
}

// ---------------- QKV GEMM: block = 128(o) x 64(l), K=256 ----------------
// All 8 waves share the B tile (y_t rows), staged async + double-buffered in LDS.
__global__ __launch_bounds__(256) void qkv_kernel(const half_t* __restrict__ wq,
                                                  const half_t* __restrict__ y_t,
                                                  const float* __restrict__ qkv_b,
                                                  half_t* __restrict__ qk,
                                                  half_t* __restrict__ vv) {
  int bx   = blockIdx.x;                 // 16 b * 6 mblk * 16 nt = 1536
  int b    = bx / 96;
  int rem  = bx - b * 96;
  int mblk = rem >> 4;
  int nt   = rem & 15;
  int l0   = nt << 6;
  int tid  = threadIdx.x, wave = tid >> 5, lane = tid & 31;
  int o0   = mblk * 128 + wave * 16;
  const half_t* yb = y_t + (size_t)b * LCNT * CH;

  __shared__ __align__(16) half_t btile[2][64][32];   // [l][k] for current k-step
  int srow = tid >> 2, sch = (tid & 3) << 3;          // 64 rows x 4 chunks of 16B
  const half_t* gsrc = yb + (size_t)(l0 + srow) * CH + sch;

  async_b128(gsrc, &btile[0][srow][sch]);             // prologue: k-step 0
  v8f acc[4] = {};
  for (int ks = 0; ks < 8; ++ks) {
    int buf = ks & 1;
    if (ks + 1 < 8) {
      async_b128(gsrc + (ks + 1) * 32, &btile[buf ^ 1][srow][sch]);
      wait_async_le<1>();                              // retire current buffer
    } else {
      wait_async_le<0>();
    }
    __syncthreads();
    v16h af = load_frag(wq, CH, o0, ks * 32);          // weights: L2-hot
#pragma unroll
    for (int j = 0; j < 4; ++j) {
      v16h bf = load_frag(&btile[buf][0][0], 32, j * 16, 0);
      acc[j] = wmma_f16(af, bf, acc[j]);
    }
    __syncthreads();                                   // protect buf before overwrite
  }

  int mh = (lane >> 4) << 3;
  float pb[8];
#pragma unroll
  for (int r = 0; r < 8; ++r) pb[r] = qkv_b[o0 + mh + r];
#pragma unroll
  for (int j = 0; j < 4; ++j) {
#pragma unroll
    for (int r = 0; r < 8; ++r) {
      int o = o0 + mh + r;
      int l = l0 + j * 16 + (lane & 15);
      float val = acc[j][r] + pb[r];
      int sec = o >> 8;                 // 0=q 1=k 2=v
      int h = (o >> 6) & 3;
      int d = o & 63;
      if (sec == 2)
        vv[(((size_t)b * NH + h) * HD + d) * LCNT + l] = (half_t)val;   // [b][h][d][l]
      else
        qk[((((size_t)b * 2 + sec) * NH + h) * LCNT + l) * HD + d] = (half_t)val; // [b][s][h][l][d]
    }
  }
}

// ---------------- flash attention: K/V tiles async-staged, shared by 8 waves ----------------
__global__ __launch_bounds__(256) void attn_kernel(const half_t* __restrict__ qk,
                                                   const half_t* __restrict__ vv,
                                                   half_t* __restrict__ ao) {
  int bh = blockIdx.x >> 3;
  int ib = blockIdx.x & 7;
  int b = bh >> 2, h = bh & 3;
  int tid = threadIdx.x, wave = tid >> 5, lane = tid & 31;
  const half_t* qt = qk + (((size_t)b * 2 * NH + h) * LCNT) * HD;
  const half_t* kt = qk + ((((size_t)b * 2 + 1) * NH + h) * LCNT) * HD;
  const half_t* vp = vv + (((size_t)b * NH + h) * HD) * LCNT;
  int i0 = ib * 128 + wave * 16;

  __shared__ __align__(16) half_t ktile[2][32][64];   // [j][d]
  __shared__ __align__(16) half_t vtile[2][64][32];   // [d][j]
  __shared__ __align__(16) half_t pstage[8][16 * 32]; // per-wave P re-layout
  half_t* ps = pstage[wave];

  int krow = tid >> 3, kch = (tid & 7) << 3;          // 32 rows x 8 chunks
  int vrow = tid >> 2, vch = (tid & 3) << 3;          // 64 rows x 4 chunks
  const half_t* kgb = kt + (size_t)krow * HD + kch;
  const half_t* vgb = vp + (size_t)vrow * LCNT + vch;

  v16h qf0 = load_frag(qt, HD, i0, 0);
  v16h qf1 = load_frag(qt, HD, i0, 32);
  float m_i[8], l_i[8];
  v8f O[4] = {};
#pragma unroll
  for (int r = 0; r < 8; ++r) { m_i[r] = -1e30f; l_i[r] = 0.f; }
  const float scale = 0.125f;   // 64^-0.5

  async_b128(kgb, &ktile[0][krow][kch]);              // prologue j-block 0
  async_b128(vgb, &vtile[0][vrow][vch]);

  for (int js = 0; js < 32; ++js) {
    int buf = js & 1;
    if (js + 1 < 32) {
      int jn = (js + 1) * 32;
      async_b128(kgb + (size_t)jn * HD, &ktile[buf ^ 1][krow][kch]);
      async_b128(vgb + jn,              &vtile[buf ^ 1][vrow][vch]);
      wait_async_le<2>();
    } else {
      wait_async_le<0>();
    }
    __syncthreads();

    v16h kf00 = load_frag(&ktile[buf][0][0], HD, 0,  0);
    v16h kf01 = load_frag(&ktile[buf][0][0], HD, 0,  32);
    v16h kf10 = load_frag(&ktile[buf][0][0], HD, 16, 0);
    v16h kf11 = load_frag(&ktile[buf][0][0], HD, 16, 32);
    v8f s0 = {}, s1 = {};
    s0 = wmma_f16(qf0, kf00, s0);
    s0 = wmma_f16(qf1, kf01, s0);
    s1 = wmma_f16(qf0, kf10, s1);
    s1 = wmma_f16(qf1, kf11, s1);

    // online softmax: row m = r + 8*(lane>=16) lives across one 16-lane half
#pragma unroll
    for (int r = 0; r < 8; ++r) {
      float a0 = s0[r] * scale, a1 = s1[r] * scale;
      float rm = fmaxf(a0, a1);
#pragma unroll
      for (int off = 1; off < 16; off <<= 1) rm = fmaxf(rm, __shfl_xor(rm, off, 32));
      float mn = fmaxf(m_i[r], rm);
      float alpha = __expf(m_i[r] - mn);
      m_i[r] = mn;
      float e0 = __expf(a0 - mn), e1 = __expf(a1 - mn);
      float rsum = e0 + e1;
#pragma unroll
      for (int off = 1; off < 16; off <<= 1) rsum += __shfl_xor(rsum, off, 32);
      l_i[r] = l_i[r] * alpha + rsum;
#pragma unroll
      for (int tt = 0; tt < 4; ++tt) O[tt][r] *= alpha;
      int row = r + ((lane >> 4) << 3);
      ps[row * 32 + (lane & 15)]      = (half_t)e0;
      ps[row * 32 + 16 + (lane & 15)] = (half_t)e1;
    }
    // same-wave DS ordering: P store -> A-fragment reload
    v16h pf = load_frag(ps, 32, 0, 0);
#pragma unroll
    for (int tt = 0; tt < 4; ++tt) {
      v16h vf = load_frag(&vtile[buf][0][0], 32, tt * 16, 0);
      O[tt] = wmma_f16(pf, vf, O[tt]);
    }
    __syncthreads();                                   // buf reusable next iter
  }

#pragma unroll
  for (int r = 0; r < 8; ++r) {
    float inv = 1.0f / l_i[r];
#pragma unroll
    for (int tt = 0; tt < 4; ++tt) O[tt][r] *= inv;
  }
#pragma unroll
  for (int tt = 0; tt < 4; ++tt)
#pragma unroll
    for (int r = 0; r < 8; ++r) {
      int i = i0 + r + ((lane >> 4) << 3);
      int c = h * HD + tt * 16 + (lane & 15);
      ao[((size_t)b * LCNT + i) * CH + c] = (half_t)(O[tt][r]);   // [b][l][c]
    }
}

// ---------------- proj GEMM + bias + residual: block = 128(o) x 64(l) ----------------
__global__ __launch_bounds__(256) void proj_kernel(const half_t* __restrict__ wp,
                                                   const half_t* __restrict__ ao,
                                                   const float* __restrict__ proj_b,
                                                   const float* __restrict__ x,
                                                   float* __restrict__ out) {
  int bx   = blockIdx.x;                 // 16 b * 2 mblk * 16 nt = 512
  int b    = bx / 32;
  int rem  = bx - b * 32;
  int mblk = rem >> 4;
  int nt   = rem & 15;
  int l0   = nt << 6;
  int tid  = threadIdx.x, wave = tid >> 5, lane = tid & 31;
  int o0   = mblk * 128 + wave * 16;
  const half_t* ab = ao + (size_t)b * LCNT * CH;

  __shared__ __align__(16) half_t btile[2][64][32];
  int srow = tid >> 2, sch = (tid & 3) << 3;
  const half_t* gsrc = ab + (size_t)(l0 + srow) * CH + sch;

  async_b128(gsrc, &btile[0][srow][sch]);
  v8f acc[4] = {};
  for (int ks = 0; ks < 8; ++ks) {
    int buf = ks & 1;
    if (ks + 1 < 8) {
      async_b128(gsrc + (ks + 1) * 32, &btile[buf ^ 1][srow][sch]);
      wait_async_le<1>();
    } else {
      wait_async_le<0>();
    }
    __syncthreads();
    v16h af = load_frag(wp, CH, o0, ks * 32);
#pragma unroll
    for (int j = 0; j < 4; ++j) {
      v16h bf = load_frag(&btile[buf][0][0], 32, j * 16, 0);
      acc[j] = wmma_f16(af, bf, acc[j]);
    }
    __syncthreads();
  }

  int mh = (lane >> 4) << 3;
  float pb[8];
#pragma unroll
  for (int r = 0; r < 8; ++r) pb[r] = proj_b[o0 + mh + r];
#pragma unroll
  for (int j = 0; j < 4; ++j)
#pragma unroll
    for (int r = 0; r < 8; ++r) {
      int o = o0 + mh + r;
      int l = l0 + j * 16 + (lane & 15);
      size_t idx = ((size_t)b * CH + o) * LCNT + l;
      out[idx] = x[idx] + pb[r] + acc[j][r];
    }
}

extern "C" void kernel_launch(void* const* d_in, const int* in_sizes, int n_in,
                              void* d_out, int out_size, void* d_ws, size_t ws_size,
                              hipStream_t stream) {
  const float* x      = (const float*)d_in[0];
  const float* gn_w   = (const float*)d_in[1];
  const float* gn_b   = (const float*)d_in[2];
  const float* qkv_w  = (const float*)d_in[3];
  const float* qkv_b  = (const float*)d_in[4];
  const float* proj_w = (const float*)d_in[5];
  const float* proj_b = (const float*)d_in[6];
  float* out = (float*)d_out;

  // workspace layout (bytes), total ~40.5 MB
  char* ws = (char*)d_ws;
  half_t* wq  = (half_t*)(ws);              // 768*256 f16
  half_t* wp  = (half_t*)(ws +   393216);   // 256*256 f16
  half_t* y_t = (half_t*)(ws +   524288);   // 16*1024*256 f16
  half_t* qk  = (half_t*)(ws +  8912896);   // 16*2*4*1024*64 f16
  half_t* vv  = (half_t*)(ws + 25690112);   // 16*4*64*1024 f16
  half_t* ao  = (half_t*)(ws + 34078720);   // 16*1024*256 f16

  cvt_kernel <<<768,  256, 0, stream>>>(qkv_w, proj_w, wq, wp);
  gn_kernel  <<<512,  256, 0, stream>>>(x, gn_w, gn_b, y_t);
  qkv_kernel <<<1536, 256, 0, stream>>>(wq, y_t, qkv_b, qk, vv);
  attn_kernel<<<512,  256, 0, stream>>>(qk, vv, ao);
  proj_kernel<<<512,  256, 0, stream>>>(wp, ao, proj_b, x, out);
}
